// CausalSelfAttention_3169685864967
// MI455X (gfx1250) — matrix operand order repaired and checked by reading, not compiled, not run on hardware
//
#include <hip/hip_runtime.h>

typedef __bf16 bf16_t;
typedef __attribute__((ext_vector_type(16))) __bf16 v16bf;
typedef __attribute__((ext_vector_type(8)))  __bf16 v8bf;
typedef __attribute__((ext_vector_type(4)))  __bf16 v4bf;
typedef __attribute__((ext_vector_type(8)))  float   v8f;
typedef __attribute__((ext_vector_type(4)))  int     v4i;
typedef unsigned int u32;

constexpr int Bb   = 4;
constexpr int Tt   = 2048;
constexpr int Cc   = 2048;
constexpr int NH   = 16;
constexpr int NKV  = 4;
constexpr int HD   = 128;
constexpr int QKVC = (NH + 2 * NKV) * HD;   // 3072
constexpr int Mrows = Bb * Tt;              // 8192

// ---------------------------------------------------------------- WMMA helpers
__device__ __forceinline__ v8f wmma_bf16(v16bf a, v16bf b, v8f c) {
  return __builtin_amdgcn_wmma_f32_16x16x32_bf16(
      false, a, false, b, (short)0, c, false, false);
}

__device__ __forceinline__ v16bf frag16(const bf16_t* lo, const bf16_t* hi) {
  v8bf a = *(const v8bf*)lo;
  v8bf b = *(const v8bf*)hi;
  v16bf r;
#pragma unroll
  for (int i = 0; i < 8; ++i) { r[i] = a[i]; r[8 + i] = b[i]; }
  return r;
}

// ---------------------------------------------------------------- CDNA5 data movers
// Flat shared pointer: addr[31:0] is the LDS byte offset (ISA 10.2 aperture rules).
__device__ __forceinline__ u32 lds_off_of(const void* p) {
  return (u32)(uintptr_t)p;
}

__device__ __forceinline__ void async_b128(const void* g, void* l) {
#if __has_builtin(__builtin_amdgcn_global_load_async_to_lds_b128)
  __builtin_amdgcn_global_load_async_to_lds_b128(
      (__attribute__((address_space(1))) v4i*)(uintptr_t)g,
      (__attribute__((address_space(3))) v4i*)(uintptr_t)l, 0, 0);
#else
  *(v8bf*)l = *(const v8bf*)g;
#endif
}

__device__ __forceinline__ void wait_async_stage() {
#if __has_builtin(__builtin_amdgcn_global_load_async_to_lds_b128)
#if __has_builtin(__builtin_amdgcn_s_wait_asynccnt)
  __builtin_amdgcn_s_wait_asynccnt(0);
#else
  asm volatile("s_wait_asynccnt 0" ::: "memory");
#endif
#endif
}

#if __has_builtin(__builtin_amdgcn_tensor_load_to_lds)
#define HAVE_TDM 1
typedef unsigned int tdm_v4u __attribute__((ext_vector_type(4)));
typedef int          tdm_v8i __attribute__((ext_vector_type(8)));
typedef int          tdm_v4i __attribute__((ext_vector_type(4)));

// 2D TDM tile load: tile_w x tile_h bf16 elements, row stride = stride elems,
// LDS rows padded +4 DWORDs every 32 DWORDs (64 bf16 row -> 72-elem LDS stride).
__device__ __forceinline__ void tdm_load_2d(const void* gsrc, u32 lds_byte,
                                            u32 tile_w, u32 tile_h, u32 stride) {
  unsigned long long ga = (unsigned long long)(uintptr_t)gsrc;
  tdm_v4u g0;
  g0[0] = 1u;                                   // count=1 (valid), user mode
  g0[1] = lds_byte;                             // lds_addr
  g0[2] = (u32)ga;                              // global_addr[31:0]
  g0[3] = ((u32)(ga >> 32) & 0x1ffffffu) | (2u << 30);  // global_addr hi | type=2
  const u32 d0 = stride;                        // tensor_dim0 (elems)
  const u32 d1 = 1u << 20;                      // tensor_dim1 (large)
  tdm_v8i g1;
  g1[0] = (int)((1u << 16)                      // data_size = 2 bytes
              | (1u << 20)                      // pad_enable
              | (4u << 22)                      // pad_interval: every 32 DWORDs
              | (3u << 25));                    // pad_amount: 4 DWORDs
  g1[1] = (int)((d0 & 0xffffu) << 16);                            // dim0 lo
  g1[2] = (int)(((d0 >> 16) & 0xffffu) | ((d1 & 0xffffu) << 16)); // dim0 hi | dim1 lo
  g1[3] = (int)(((d1 >> 16) & 0xffffu) | (tile_w << 16));         // dim1 hi | tile_dim0
  g1[4] = (int)(tile_h & 0xffffu);                                // tile_dim1 (tile_dim2=0)
  g1[5] = (int)stride;                                            // dim0_stride lo
  g1[6] = 0;
  g1[7] = 0;
  tdm_v4i gz; gz[0] = 0; gz[1] = 0; gz[2] = 0; gz[3] = 0;
#if __has_include(<hip/amd_detail/amd_gfx1250_TDM.h>)
  tdm_v8i gz8;
#pragma unroll
  for (int i = 0; i < 8; ++i) gz8[i] = 0;
  __builtin_amdgcn_tensor_load_to_lds(g0, g1, gz, gz, gz8, 0);
#else
  __builtin_amdgcn_tensor_load_to_lds(g0, g1, gz, gz, 0);
#endif
}

__device__ __forceinline__ void wait_tensor0() {
#if __has_builtin(__builtin_amdgcn_s_wait_tensorcnt)
  __builtin_amdgcn_s_wait_tensorcnt(0);
#else
  asm volatile("s_wait_tensorcnt 0" ::: "memory");
#endif
}
#endif  // HAVE_TDM

// ---------------------------------------------------------------- fp32 -> bf16
__global__ void cvt_f32_to_bf16(const float* __restrict__ in,
                                bf16_t* __restrict__ out, int n) {
  int i = (blockIdx.x * blockDim.x + threadIdx.x) * 4;
  int stride = gridDim.x * blockDim.x * 4;
  for (; i < n; i += stride) {
    float4 f = *(const float4*)(in + i);
    v4bf o;
    o[0] = (bf16_t)f.x; o[1] = (bf16_t)f.y;
    o[2] = (bf16_t)f.z; o[3] = (bf16_t)f.w;
    *(v4bf*)(out + i) = o;
  }
}

// ---------------------------------------------------------------- NT GEMM
// C[M,N] = A[M,K] * B[N,K]^T, bf16 in / OutT out.
// 256 thr = 8 waves; block tile 128x128, BK=64, double-buffered LDS staged by
// TDM (wave 0, one descriptor per matrix tile) or async-to-LDS fallback.
template <typename OutT>
__global__ __launch_bounds__(256, 2) void gemm_bf16_nt(
    const bf16_t* __restrict__ A, const bf16_t* __restrict__ B,
    OutT* __restrict__ C, int M, int N, int K) {
  constexpr int BKP = 72;                       // 64 + 8 pad (144B row stride)
  __shared__ bf16_t As[2][128][BKP];
  __shared__ bf16_t Bs[2][128][BKP];

  const int bm   = blockIdx.y * 128;
  const int bn   = blockIdx.x * 128;
  const int tid  = threadIdx.x;
  const int wave = tid >> 5;
  const int lane = tid & 31;
  const int l15  = lane & 15;
  const int kh   = lane >> 4;
  const int wm   = (wave & 3) * 32;
  const int wn   = (wave >> 2) * 64;

  v8f acc[2][4] = {};

  const bf16_t* Ap = A + (size_t)bm * K;
  const bf16_t* Bp = B + (size_t)bn * K;

  auto stage = [&](int buf, int k0) {
#ifdef HAVE_TDM
    if (wave == 0) {
      tdm_load_2d(Ap + k0, lds_off_of(&As[buf][0][0]), 64u, 128u, (u32)K);
      tdm_load_2d(Bp + k0, lds_off_of(&Bs[buf][0][0]), 64u, 128u, (u32)K);
    }
#else
#pragma unroll
    for (int c = 0; c < 4; ++c) {
      const int id  = c * 256 + tid;            // 0..1023 chunks of 16B
      const int row = id >> 3;                  // 0..127
      const int col = (id & 7) * 8;             // 0..56
      async_b128(Ap + (size_t)row * K + k0 + col, &As[buf][row][col]);
      async_b128(Bp + (size_t)row * K + k0 + col, &Bs[buf][row][col]);
    }
#endif
  };
  auto stage_wait = [&]() {
#ifdef HAVE_TDM
    if (wave == 0) wait_tensor0();
#else
    wait_async_stage();
#endif
  };

  stage(0, 0);
  stage_wait();
  __syncthreads();

  int buf = 0;
  for (int k0 = 0; k0 < K; k0 += 64) {
    if (k0 + 64 < K) stage(buf ^ 1, k0 + 64);

#pragma unroll
    for (int ks = 0; ks < 2; ++ks) {
      v16bf af[2], bfr[4];
#pragma unroll
      for (int i = 0; i < 2; ++i)
        af[i] = frag16(&As[buf][wm + i * 16 + l15][ks * 32 + kh * 8],
                       &As[buf][wm + i * 16 + l15][ks * 32 + 16 + kh * 8]);
#pragma unroll
      for (int j = 0; j < 4; ++j)
        bfr[j] = frag16(&Bs[buf][wn + j * 16 + l15][ks * 32 + kh * 8],
                        &Bs[buf][wn + j * 16 + l15][ks * 32 + 16 + kh * 8]);
#pragma unroll
      for (int i = 0; i < 2; ++i)
#pragma unroll
        for (int j = 0; j < 4; ++j)
          acc[i][j] = wmma_bf16(af[i], bfr[j], acc[i][j]);
    }

    stage_wait();      // my staging for buf^1 complete
    __syncthreads();   // everyone done reading buf; buf^1 valid for all
    buf ^= 1;
  }

#pragma unroll
  for (int i = 0; i < 2; ++i) {
#pragma unroll
    for (int j = 0; j < 4; ++j) {
      const int col = bn + wn + j * 16 + l15;
#pragma unroll
      for (int v = 0; v < 8; ++v) {
        const int row = bm + wm + i * 16 + kh * 8 + v;
        C[(size_t)row * N + col] = (OutT)acc[i][j][v];
      }
    }
  }
}

// ---------------------------------------------------------------- RoPE (in place on q,k)
__global__ void rope_kernel(bf16_t* __restrict__ qkv) {
  const int total = Mrows * (NH + NKV) * (HD / 2);
  int idx = blockIdx.x * blockDim.x + threadIdx.x;
  if (idx >= total) return;
  const int pair = idx & 63;
  int tmp = idx >> 6;
  const int head = tmp % (NH + NKV);
  const int row  = tmp / (NH + NKV);
  const int t    = row & (Tt - 1);
  bf16_t* p = qkv + (size_t)row * QKVC + head * HD + 2 * pair;
  const float x1 = (float)p[0];
  const float x2 = (float)p[1];
  const float inv = __powf(10000.0f, -(float)(2 * pair) / (float)HD);
  float s, c;
  __sincosf((float)t * inv, &s, &c);
  p[0] = (bf16_t)(x1 * c - x2 * s);
  p[1] = (bf16_t)(x1 * s + x2 * c);
}

// ---------------------------------------------------------------- flash attention
// grid (T/128, NH, B); 256 thr = 8 waves; wave owns 16 q rows x full HD=128.
__global__ __launch_bounds__(256, 1) void attn_kernel(
    const bf16_t* __restrict__ qkv, bf16_t* __restrict__ y) {
  __shared__ bf16_t Ks[32][136];          // kv x d (+8 pad)
  __shared__ bf16_t Vt[128][40];          // d x kv transposed (+8 pad)
  __shared__ bf16_t Ps[8][16][40];        // per-wave P scratch

  const int tid  = threadIdx.x;
  const int wave = tid >> 5;
  const int lane = tid & 31;
  const int l15  = lane & 15;
  const int kh   = lane >> 4;
  const int kh8  = kh * 8;

  const int b     = blockIdx.z;
  const int h     = blockIdx.y;
  const int kvh   = h >> 2;
  const int q0blk = blockIdx.x * 128;
  const int q0w   = q0blk + wave * 16;
  const size_t bT0 = (size_t)b * Tt;

  const bf16_t* qp = qkv + (bT0 + q0w + l15) * QKVC + (size_t)h * HD;
  v16bf qf[4];
#pragma unroll
  for (int c = 0; c < 4; ++c)
    qf[c] = frag16(qp + c * 32 + kh8, qp + c * 32 + 16 + kh8);

  v8f o[8] = {};
  float m_[8], l_[8], alpha[8];
#pragma unroll
  for (int v = 0; v < 8; ++v) { m_[v] = -1e30f; l_[v] = 0.f; }

  const float sc = 0.08838834764831845f;  // 1/sqrt(128)
  const int j_end = q0blk + 128;

  for (int j0 = 0; j0 < j_end; j0 += 32) {
    __syncthreads();
    // K tile: async DMA straight into LDS; V tile: sync load + transpose store.
#pragma unroll
    for (int it = 0; it < 2; ++it) {
      const int cidx = tid + it * 256;      // 0..511
      const int r    = cidx >> 4;           // kv row
      const int ch   = (cidx & 15) * 8;     // d chunk
      const bf16_t* gk = qkv + (bT0 + j0 + r) * QKVC + NH * HD + kvh * HD + ch;
      async_b128(gk, &Ks[r][ch]);
      v8bf vv = *(const v8bf*)(gk + NKV * HD);
#pragma unroll
      for (int e = 0; e < 8; ++e) Vt[ch + e][r] = vv[e];
    }
    wait_async_stage();
    __syncthreads();

    v8f s0 = {}, s1 = {};
#pragma unroll
    for (int c = 0; c < 4; ++c) {
      v16bf k0f = frag16(&Ks[l15][c * 32 + kh * 8],
                         &Ks[l15][c * 32 + 16 + kh * 8]);
      v16bf k1f = frag16(&Ks[16 + l15][c * 32 + kh * 8],
                         &Ks[16 + l15][c * 32 + 16 + kh * 8]);
      s0 = wmma_bf16(qf[c], k0f, s0);
      s1 = wmma_bf16(qf[c], k1f, s1);
    }

#pragma unroll
    for (int v = 0; v < 8; ++v) {
      const int qi = q0w + kh8 + v;
      float a0 = s0[v] * sc;
      float a1 = s1[v] * sc;
      if (j0 + l15 > qi)      a0 = -1e30f;
      if (j0 + 16 + l15 > qi) a1 = -1e30f;
      float rm = fmaxf(a0, a1);
#pragma unroll
      for (int off = 1; off < 16; off <<= 1)
        rm = fmaxf(rm, __shfl_xor(rm, off, 32));
      const float mn = fmaxf(m_[v], rm);
      const float al = __expf(m_[v] - mn);
      const float p0 = __expf(a0 - mn);
      const float p1 = __expf(a1 - mn);
      float rs = p0 + p1;
#pragma unroll
      for (int off = 1; off < 16; off <<= 1)
        rs += __shfl_xor(rs, off, 32);
      l_[v] = l_[v] * al + rs;
      m_[v] = mn;
      alpha[v] = al;
      Ps[wave][kh8 + v][l15]      = (bf16_t)p0;
      Ps[wave][kh8 + v][16 + l15] = (bf16_t)p1;
    }

#pragma unroll
    for (int d = 0; d < 8; ++d)
#pragma unroll
      for (int v = 0; v < 8; ++v) o[d][v] *= alpha[v];

    v16bf pf = frag16(&Ps[wave][l15][kh8], &Ps[wave][l15][16 + kh8]);
#pragma unroll
    for (int d = 0; d < 8; ++d) {
      v16bf vf = frag16(&Vt[d * 16 + l15][kh * 8],
                        &Vt[d * 16 + l15][16 + kh * 8]);
      o[d] = wmma_bf16(pf, vf, o[d]);
    }
  }

#pragma unroll
  for (int d = 0; d < 8; ++d) {
#pragma unroll
    for (int v = 0; v < 8; ++v) {
      const size_t row = bT0 + q0w + kh8 + v;
      const int col = h * HD + d * 16 + l15;
      y[row * Cc + col] = (bf16_t)(o[d][v] / l_[v]);
    }
  }
}

// ---------------------------------------------------------------- launcher
extern "C" void kernel_launch(void* const* d_in, const int* in_sizes, int n_in,
                              void* d_out, int out_size, void* d_ws, size_t ws_size,
                              hipStream_t stream) {
  (void)in_sizes; (void)n_in; (void)out_size; (void)ws_size;
  const float* x      = (const float*)d_in[0];
  const float* w_attn = (const float*)d_in[1];
  const float* w_proj = (const float*)d_in[2];
  float* out = (float*)d_out;

  char* ws = (char*)d_ws;
  size_t off = 0;
  bf16_t* xb  = (bf16_t*)(ws + off); off += (size_t)Mrows * Cc * 2;
  bf16_t* wab = (bf16_t*)(ws + off); off += (size_t)QKVC * Cc * 2;
  bf16_t* wpb = (bf16_t*)(ws + off); off += (size_t)Cc * Cc * 2;
  bf16_t* qkv = (bf16_t*)(ws + off); off += (size_t)Mrows * QKVC * 2;
  bf16_t* yb  = (bf16_t*)(ws + off); off += (size_t)Mrows * Cc * 2;

  cvt_f32_to_bf16<<<4096, 256, 0, stream>>>(x, xb, Mrows * Cc);
  cvt_f32_to_bf16<<<4096, 256, 0, stream>>>(w_attn, wab, QKVC * Cc);
  cvt_f32_to_bf16<<<4096, 256, 0, stream>>>(w_proj, wpb, Cc * Cc);

  dim3 g1(QKVC / 128, Mrows / 128);
  gemm_bf16_nt<bf16_t><<<g1, 256, 0, stream>>>(xb, wab, qkv, Mrows, QKVC, Cc);

  const int rope_total = Mrows * (NH + NKV) * (HD / 2);
  rope_kernel<<<(rope_total + 255) / 256, 256, 0, stream>>>(qkv);

  dim3 ga(Tt / 128, NH, Bb);
  attn_kernel<<<ga, 256, 0, stream>>>(qkv, yb);

  dim3 g2(Cc / 128, Mrows / 128);
  gemm_bf16_nt<float><<<g2, 256, 0, stream>>>(yb, wpb, out, Mrows, Cc, Cc);
}
